// QFormerKVCompressor_44693429682231
// MI455X (gfx1250) — matrix operand order repaired and checked by reading, not compiled, not run on hardware
//
#include <hip/hip_runtime.h>
#include <math.h>

// ---------------- types ----------------
typedef __bf16 bf16;
typedef __attribute__((ext_vector_type(16))) __bf16 v16bf;
typedef __attribute__((ext_vector_type(8)))  __bf16 v8bf;
typedef __attribute__((ext_vector_type(8)))  float  v8f;
typedef __attribute__((vector_size(16)))     int    v4i_vs;   // matches builtin param pointee

// ---------------- problem constants (match reference harness) ----------------
#define L_C   32
#define DOC_C 2048
#define H_C   4096
#define A_C   128
#define FFN_C 2048
#define KVH_C 8
#define HD_C  128
#define KV_C  (KVH_C * HD_C)   // 1024
#define NQ_C  512              // DOC / compression_ratio(4), capped at 2048

// gfx1250 async global->LDS path (ASYNCcnt), with compile-safe fallback
#if defined(__gfx1250__) && __has_builtin(__builtin_amdgcn_global_load_async_to_lds_b128) && __has_builtin(__builtin_amdgcn_s_wait_asynccnt)
#define HAS_ASYNC_LDS 1
#define GAS1 __attribute__((address_space(1)))
#define LAS3 __attribute__((address_space(3)))
#else
#define HAS_ASYNC_LDS 0
#endif

__device__ __forceinline__ bf16 f2bf(float f) {
  unsigned u = __builtin_bit_cast(unsigned, f);
  unsigned r = u + 0x7FFFu + ((u >> 16) & 1u);   // round-to-nearest-even
  unsigned short h = (unsigned short)(r >> 16);
  return __builtin_bit_cast(bf16, h);
}

// ---------------- elementwise convert f32 -> bf16 ----------------
__global__ void k_f32_to_bf16(const float* __restrict__ s, bf16* __restrict__ d, size_t n) {
  size_t i = (size_t)blockIdx.x * blockDim.x + threadIdx.x;
  size_t stride = (size_t)gridDim.x * blockDim.x;
  for (; i < n; i += stride) d[i] = f2bf(s[i]);
}

// ---------------- per-layer transpose: hs[D][H] f32 -> hsT[H][D] bf16 ----------------
__global__ __launch_bounds__(256)
void k_transpose_bf(const float* __restrict__ src, bf16* __restrict__ dst) {
  __shared__ float tile[32][33];
  const int l = blockIdx.z;
  src += (long long)l * DOC_C * H_C;
  dst += (long long)l * DOC_C * H_C;
  const int d0 = blockIdx.y * 32;
  const int h0 = blockIdx.x * 32;
  const int tx = threadIdx.x & 31;
  const int ty = threadIdx.x >> 5;            // 8 rows / iter
  #pragma unroll
  for (int r = ty; r < 32; r += 8)
    tile[r][tx] = src[(long long)(d0 + r) * H_C + (h0 + tx)];
  __syncthreads();
  #pragma unroll
  for (int r = ty; r < 32; r += 8)
    dst[(long long)(h0 + r) * DOC_C + (d0 + tx)] = f2bf(tile[tx][r]);
}

// ---------------- mean-pool queries: hs[L,D,H] -> q[L,NQ,H] (f32 + bf16) ----------------
__global__ void k_pool(const float* __restrict__ hs, const int* __restrict__ ratio_p,
                       float* __restrict__ qf, bf16* __restrict__ qb) {
  const int ratio = *ratio_p;
  const float inv = 1.0f / (float)ratio;
  size_t i = (size_t)blockIdx.x * blockDim.x + threadIdx.x;
  size_t stride = (size_t)gridDim.x * blockDim.x;
  const size_t total = (size_t)L_C * NQ_C * H_C;
  for (; i < total; i += stride) {
    size_t h = i % H_C;
    size_t t = i / H_C;
    size_t q = t % NQ_C;
    size_t l = t / NQ_C;
    const float* base = hs + ((size_t)l * DOC_C + q * ratio) * H_C + h;
    float s = 0.f;
    for (int j = 0; j < ratio; ++j) s += base[(size_t)j * H_C];
    float m = s * inv;
    qf[i] = m;
    qb[i] = f2bf(m);
  }
}

// ---------------- generic batched NT GEMM: C[M,N] = alpha * A[M,K] * B[N,K]^T + bias[N] ----
// bf16 inputs, fp32 accumulate via v_wmma_f32_16x16x32_bf16.
// Block tile 64x64, 4 waves (wave32), each wave = 32x32 = 2x2 WMMA tiles, BK=32.
// Staging uses GLOBAL_LOAD_ASYNC_TO_LDS_B128 (ASYNCcnt) when available.
#define TM 64
#define TN 64
#define TK 32
#define LDP (TK + 8)   // LDS row pitch (bf16 elems), 16B-aligned

__global__ __launch_bounds__(128)
void k_gemm(const bf16* __restrict__ A, const bf16* __restrict__ B,
            float* __restrict__ Cf, bf16* __restrict__ Cb,
            const float* __restrict__ bias,
            int M, int N, int K,
            long long sAb, long long sBb, long long sCb, long long sBiasb,
            float alpha) {
  __shared__ __align__(16) bf16 ldsA[TM][LDP];
  __shared__ __align__(16) bf16 ldsB[TN][LDP];

  const int l = blockIdx.z;
  A += (long long)l * sAb;
  B += (long long)l * sBb;
  if (Cf)   Cf   += (long long)l * sCb;
  if (Cb)   Cb   += (long long)l * sCb;
  if (bias) bias += (long long)l * sBiasb;

  const int bm = blockIdx.y * TM;
  const int bn = blockIdx.x * TN;
  const int tid  = threadIdx.x;
  const int lane = tid & 31;
  const int wave = tid >> 5;
  const int wm = (wave >> 1) * 32;  // wave row offset in block tile
  const int wn = (wave & 1) * 32;  // wave col offset
  const int lr = lane & 15;
  const int hi = lane >> 4;

  // branch-free staging coordinates: thread handles rows sr and sr+32 at column sc
  const int sr = tid >> 2;
  const int sc = (tid & 3) * 8;

  v8f acc[2][2] = {};

  for (int k0 = 0; k0 < K; k0 += TK) {
#if HAS_ASYNC_LDS
    __builtin_amdgcn_global_load_async_to_lds_b128(
        (GAS1 v4i_vs*)&A[(long long)(bm + sr) * K + k0 + sc],
        (LAS3 v4i_vs*)&ldsA[sr][sc], 0, 0);
    __builtin_amdgcn_global_load_async_to_lds_b128(
        (GAS1 v4i_vs*)&A[(long long)(bm + sr + 32) * K + k0 + sc],
        (LAS3 v4i_vs*)&ldsA[sr + 32][sc], 0, 0);
    __builtin_amdgcn_global_load_async_to_lds_b128(
        (GAS1 v4i_vs*)&B[(long long)(bn + sr) * K + k0 + sc],
        (LAS3 v4i_vs*)&ldsB[sr][sc], 0, 0);
    __builtin_amdgcn_global_load_async_to_lds_b128(
        (GAS1 v4i_vs*)&B[(long long)(bn + sr + 32) * K + k0 + sc],
        (LAS3 v4i_vs*)&ldsB[sr + 32][sc], 0, 0);
#else
    *(v8bf*)&ldsA[sr][sc]      = *(const v8bf*)&A[(long long)(bm + sr) * K + k0 + sc];
    *(v8bf*)&ldsA[sr + 32][sc] = *(const v8bf*)&A[(long long)(bm + sr + 32) * K + k0 + sc];
    *(v8bf*)&ldsB[sr][sc]      = *(const v8bf*)&B[(long long)(bn + sr) * K + k0 + sc];
    *(v8bf*)&ldsB[sr + 32][sc] = *(const v8bf*)&B[(long long)(bn + sr + 32) * K + k0 + sc];
#endif
    if (k0 + TK < K) {   // prefetch next K tile (global_prefetch_b8)
      __builtin_prefetch(&A[(long long)(bm + (tid >> 1)) * K + k0 + TK], 0, 1);
      __builtin_prefetch(&B[(long long)(bn + (tid >> 1)) * K + k0 + TK], 0, 1);
    }
#if HAS_ASYNC_LDS
    __builtin_amdgcn_s_wait_asynccnt(0);
#endif
    __syncthreads();

    // fragment assembly per documented CDNA5 VGPR layouts
    union F { v16bf v; v8bf h[2]; };
    F af[2], bfr[2];
    const int ab = hi * 8;    // A: two 8-elem K runs {ab..ab+7, 16+ab..16+ab+7}
    const int bk = hi * 8;    // B: same K runs as A {bk..bk+7, 16+bk..16+bk+7}
    #pragma unroll
    for (int i = 0; i < 2; ++i) {
      const int row = wm + i * 16 + lr;
      af[i].h[0] = *(const v8bf*)&ldsA[row][ab];
      af[i].h[1] = *(const v8bf*)&ldsA[row][16 + ab];
    }
    #pragma unroll
    for (int j = 0; j < 2; ++j) {
      const int row = wn + j * 16 + lr;
      bfr[j].h[0] = *(const v8bf*)&ldsB[row][bk];
      bfr[j].h[1] = *(const v8bf*)&ldsB[row][16 + bk];
    }
    #pragma unroll
    for (int i = 0; i < 2; ++i)
      #pragma unroll
      for (int j = 0; j < 2; ++j)
        acc[i][j] = __builtin_amdgcn_wmma_f32_16x16x32_bf16(
            false, af[i].v, false, bfr[j].v, (short)0, acc[i][j], false, false);
    __syncthreads();
  }

  // epilogue: C VGPR v holds M = v + 8*hi, N = lane&15 within each 16x16 tile
  if (Cf) {
    #pragma unroll
    for (int i = 0; i < 2; ++i)
      #pragma unroll
      for (int j = 0; j < 2; ++j) {
        const int n = bn + wn + j * 16 + lr;
        const float bv = bias ? bias[n] : 0.0f;
        #pragma unroll
        for (int v = 0; v < 8; ++v) {
          const int m = bm + wm + i * 16 + v + 8 * hi;
          Cf[(long long)m * N + n] = alpha * acc[i][j][v] + bv;
        }
      }
  }
  if (Cb) {
    #pragma unroll
    for (int i = 0; i < 2; ++i)
      #pragma unroll
      for (int j = 0; j < 2; ++j) {
        const int n = bn + wn + j * 16 + lr;
        const float bv = bias ? bias[n] : 0.0f;
        #pragma unroll
        for (int v = 0; v < 8; ++v) {
          const int m = bm + wm + i * 16 + v + 8 * hi;
          Cb[(long long)m * N + n] = f2bf(alpha * acc[i][j][v] + bv);
        }
      }
  }
}

// ---------------- softmax over D with relative-position bias, out bf16 ----------------
__global__ __launch_bounds__(256)
void k_softmax(const float* __restrict__ logits, bf16* __restrict__ attnw,
               const float* __restrict__ slope_p) {
  const int q = blockIdx.x, l = blockIdx.y;
  const long long base = ((long long)l * NQ_C + q) * DOC_C;
  const float slope = *slope_p;
  const float qpos = (q + 0.5f) / (float)NQ_C;
  float v[DOC_C / 256];
  float mx = -3.0e38f;
  #pragma unroll
  for (int it = 0; it < DOC_C / 256; ++it) {
    const int d = threadIdx.x + it * 256;
    const float kpos = (d + 0.5f) / (float)DOC_C;
    const float x = logits[base + d] - slope * fabsf(qpos - kpos);
    v[it] = x;
    mx = fmaxf(mx, x);
  }
  __shared__ float red[256];
  red[threadIdx.x] = mx; __syncthreads();
  for (int s = 128; s > 0; s >>= 1) {
    if (threadIdx.x < s) red[threadIdx.x] = fmaxf(red[threadIdx.x], red[threadIdx.x + s]);
    __syncthreads();
  }
  mx = red[0]; __syncthreads();
  float sum = 0.f;
  #pragma unroll
  for (int it = 0; it < DOC_C / 256; ++it) { v[it] = __expf(v[it] - mx); sum += v[it]; }
  red[threadIdx.x] = sum; __syncthreads();
  for (int s = 128; s > 0; s >>= 1) {
    if (threadIdx.x < s) red[threadIdx.x] += red[threadIdx.x + s];
    __syncthreads();
  }
  const float inv = 1.0f / red[0];
  #pragma unroll
  for (int it = 0; it < DOC_C / 256; ++it)
    attnw[base + threadIdx.x + it * 256] = f2bf(v[it] * inv);
}

// ---------------- x = g*attn + (1-g)*queries (f32 + bf16 outs) ----------------
__global__ void k_gated_res(const float* __restrict__ attn, const float* __restrict__ qf,
                            const float* __restrict__ g_p,
                            float* __restrict__ xf, bf16* __restrict__ xb, size_t n) {
  const float g = 1.0f / (1.0f + __expf(-*g_p));
  size_t i = (size_t)blockIdx.x * blockDim.x + threadIdx.x;
  size_t stride = (size_t)gridDim.x * blockDim.x;
  for (; i < n; i += stride) {
    const float x = g * attn[i] + (1.0f - g) * qf[i];
    xf[i] = x;
    xb[i] = f2bf(x);
  }
}

// ---------------- prod = silu(gate) * up  (bf16 out) ----------------
__global__ void k_swiglu(const float* __restrict__ gate, const float* __restrict__ up,
                         bf16* __restrict__ prod, size_t n) {
  size_t i = (size_t)blockIdx.x * blockDim.x + threadIdx.x;
  size_t stride = (size_t)gridDim.x * blockDim.x;
  for (; i < n; i += stride) {
    const float gv = gate[i];
    const float s = gv / (1.0f + __expf(-gv));
    prod[i] = f2bf(s * up[i]);
  }
}

// ---------------- x2 = sigmoid(gp)*ffn_out + x  (f32 out) ----------------
__global__ void k_ffn_res(const float* __restrict__ ffn, const float* __restrict__ x,
                          const float* __restrict__ g_p, float* __restrict__ x2, size_t n) {
  const float g = 1.0f / (1.0f + __expf(-*g_p));
  size_t i = (size_t)blockIdx.x * blockDim.x + threadIdx.x;
  size_t stride = (size_t)gridDim.x * blockDim.x;
  for (; i < n; i += stride) x2[i] = g * ffn[i] + x[i];
}

// ---------------- RMSNorm over H with per-layer weight, bf16 out ----------------
__global__ __launch_bounds__(256)
void k_rmsnorm(const float* __restrict__ x, const float* __restrict__ lnw,
               bf16* __restrict__ xn) {
  const int q = blockIdx.x, l = blockIdx.y;
  const long long base = ((long long)l * NQ_C + q) * H_C;
  float ss = 0.f;
  for (int h = threadIdx.x; h < H_C; h += 256) { const float t = x[base + h]; ss += t * t; }
  __shared__ float red[256];
  red[threadIdx.x] = ss; __syncthreads();
  for (int s = 128; s > 0; s >>= 1) {
    if (threadIdx.x < s) red[threadIdx.x] += red[threadIdx.x + s];
    __syncthreads();
  }
  const float r = rsqrtf(red[0] / (float)H_C + 1e-5f);
  for (int h = threadIdx.x; h < H_C; h += 256)
    xn[base + h] = f2bf(x[base + h] * r * lnw[(long long)l * H_C + h]);
}

// ---------------- pack [L,NQ,KV] -> [L,KVH,NQ,HD] for k and v ----------------
__global__ void k_pack_out(const float* __restrict__ kf, const float* __restrict__ vf,
                           float* __restrict__ out) {
  const size_t total = (size_t)L_C * KVH_C * NQ_C * HD_C;
  size_t i = (size_t)blockIdx.x * blockDim.x + threadIdx.x;
  size_t stride = (size_t)gridDim.x * blockDim.x;
  for (; i < total; i += stride) {
    size_t hd = i % HD_C;
    size_t t  = i / HD_C;
    size_t q  = t % NQ_C;
    t /= NQ_C;
    size_t kh = t % KVH_C;
    size_t l  = t / KVH_C;
    const size_t src = ((size_t)l * NQ_C + q) * KV_C + kh * HD_C + hd;
    out[i] = kf[src];
    out[i + total] = vf[src];
  }
}

// ---------------- host orchestration ----------------
static inline unsigned ew_blocks(size_t n) {
  size_t b = (n + 1023) / 1024;
  if (b > 262144) b = 262144;
  return (unsigned)b;
}

extern "C" void kernel_launch(void* const* d_in, const int* in_sizes, int n_in,
                              void* d_out, int out_size, void* d_ws, size_t ws_size,
                              hipStream_t stream) {
  (void)in_sizes; (void)n_in; (void)out_size; (void)ws_size;
  const float* hs    = (const float*)d_in[0];
  const float* lemb  = (const float*)d_in[1];
  const float* Wq    = (const float*)d_in[2];
  const float* Wk    = (const float*)d_in[3];
  const float* slope = (const float*)d_in[4];
  const float* rgate = (const float*)d_in[5];
  const float* gw    = (const float*)d_in[6];
  const float* uw    = (const float*)d_in[7];
  const float* dw    = (const float*)d_in[8];
  const float* frg   = (const float*)d_in[9];
  const float* kproj = (const float*)d_in[10];
  const float* vproj = (const float*)d_in[11];
  const float* lnw   = (const float*)d_in[12];
  const int*   ratio = (const int*)d_in[13];
  float* out = (float*)d_out;

  char* p = (char*)d_ws;
  auto alloc = [&](size_t bytes) -> void* {
    void* r = (void*)p;
    p += (bytes + 255) & ~(size_t)255;
    return r;
  };

  const size_t nHS  = (size_t)L_C * DOC_C * H_C;
  const size_t nQH  = (size_t)L_C * NQ_C * H_C;
  const size_t nQF  = (size_t)L_C * NQ_C * FFN_C;
  const size_t nQD  = (size_t)L_C * NQ_C * DOC_C;
  const size_t nQKV = (size_t)L_C * NQ_C * KV_C;

  bf16*  hs_bf    = (bf16*)alloc(nHS * 2);
  bf16*  hsT_bf   = (bf16*)alloc(nHS * 2);
  bf16*  wq_bf    = (bf16*)alloc((size_t)A_C * H_C * 2);
  bf16*  wk_bf    = (bf16*)alloc((size_t)A_C * H_C * 2);
  bf16*  gw_bf    = (bf16*)alloc((size_t)FFN_C * H_C * 2);
  bf16*  uw_bf    = (bf16*)alloc((size_t)FFN_C * H_C * 2);
  bf16*  dw_bf    = (bf16*)alloc((size_t)H_C * FFN_C * 2);
  bf16*  kproj_bf = (bf16*)alloc((size_t)L_C * KV_C * H_C * 2);
  bf16*  vproj_bf = (bf16*)alloc((size_t)L_C * KV_C * H_C * 2);
  float* q_f      = (float*)alloc(nQH * 4);
  bf16*  q_bfp    = (bf16*)alloc(nQH * 2);
  bf16*  qa_bf    = (bf16*)alloc((size_t)L_C * NQ_C * A_C * 2);
  bf16*  ka_bf    = (bf16*)alloc((size_t)L_C * DOC_C * A_C * 2);
  float* logits_f = (float*)alloc(nQD * 4);
  bf16*  attnw_bf = (bf16*)alloc(nQD * 2);
  float* attno_f  = (float*)alloc(nQH * 4);
  float* x_f      = (float*)alloc(nQH * 4);
  bf16*  x_bf     = (bf16*)alloc(nQH * 2);
  float* gate_f   = (float*)alloc(nQF * 4);
  float* up_f     = (float*)alloc(nQF * 4);
  bf16*  prod_bf  = (bf16*)alloc(nQF * 2);
  float* ffn_f    = (float*)alloc(nQH * 4);
  float* x2_f     = (float*)alloc(nQH * 4);
  bf16*  xn_bf    = (bf16*)alloc(nQH * 2);
  float* kout_f   = (float*)alloc(nQKV * 4);
  float* vout_f   = (float*)alloc(nQKV * 4);

  // 1) precision conversion of activations + weights (one pass each)
  k_f32_to_bf16<<<ew_blocks(nHS), 256, 0, stream>>>(hs, hs_bf, nHS);
  {
    dim3 g(H_C / 32, DOC_C / 32, L_C);
    k_transpose_bf<<<g, 256, 0, stream>>>(hs, hsT_bf);
  }
  k_f32_to_bf16<<<ew_blocks((size_t)A_C * H_C), 256, 0, stream>>>(Wq, wq_bf, (size_t)A_C * H_C);
  k_f32_to_bf16<<<ew_blocks((size_t)A_C * H_C), 256, 0, stream>>>(Wk, wk_bf, (size_t)A_C * H_C);
  k_f32_to_bf16<<<ew_blocks((size_t)FFN_C * H_C), 256, 0, stream>>>(gw, gw_bf, (size_t)FFN_C * H_C);
  k_f32_to_bf16<<<ew_blocks((size_t)FFN_C * H_C), 256, 0, stream>>>(uw, uw_bf, (size_t)FFN_C * H_C);
  k_f32_to_bf16<<<ew_blocks((size_t)H_C * FFN_C), 256, 0, stream>>>(dw, dw_bf, (size_t)H_C * FFN_C);
  k_f32_to_bf16<<<ew_blocks((size_t)L_C * KV_C * H_C), 256, 0, stream>>>(kproj, kproj_bf, (size_t)L_C * KV_C * H_C);
  k_f32_to_bf16<<<ew_blocks((size_t)L_C * KV_C * H_C), 256, 0, stream>>>(vproj, vproj_bf, (size_t)L_C * KV_C * H_C);

  // 2) mean-pool queries
  k_pool<<<ew_blocks(nQH), 256, 0, stream>>>(hs, ratio, q_f, q_bfp);

  auto gemm = [&](const bf16* Ab, const bf16* Bb, float* Cf, bf16* Cb, const float* bias,
                  int M, int N, int K,
                  long long sA, long long sB, long long sC, long long sBias, float alpha) {
    dim3 g(N / TN, M / TM, L_C);
    k_gemm<<<g, 128, 0, stream>>>(Ab, Bb, Cf, Cb, bias, M, N, K, sA, sB, sC, sBias, alpha);
  };

  const float inv_sqrt_a = 0.08838834764831845f;  // 1/sqrt(A_C)

  // 3) k = hs @ Wk^T            [D,A] per layer
  gemm(hs_bf, wk_bf, nullptr, ka_bf, nullptr,
       DOC_C, A_C, H_C, (long long)DOC_C * H_C, 0, (long long)DOC_C * A_C, 0, 1.0f);
  // 4) q = queries @ Wq^T + layer_emb   [NQ,A]
  gemm(q_bfp, wq_bf, nullptr, qa_bf, lemb,
       NQ_C, A_C, H_C, (long long)NQ_C * H_C, 0, (long long)NQ_C * A_C, A_C, 1.0f);
  // 5) logits = scale * q @ k^T          [NQ,D]
  gemm(qa_bf, ka_bf, logits_f, nullptr, nullptr,
       NQ_C, DOC_C, A_C, (long long)NQ_C * A_C, (long long)DOC_C * A_C,
       (long long)NQ_C * DOC_C, 0, inv_sqrt_a);
  // 6) softmax(+pos bias) -> attn_w bf16
  {
    dim3 g(NQ_C, L_C);
    k_softmax<<<g, 256, 0, stream>>>(logits_f, attnw_bf, slope);
  }
  // 7) attn_out = attn_w @ hs   (via hsT, NT)   [NQ,H]
  gemm(attnw_bf, hsT_bf, attno_f, nullptr, nullptr,
       NQ_C, H_C, DOC_C, (long long)NQ_C * DOC_C, (long long)H_C * DOC_C,
       (long long)NQ_C * H_C, 0, 1.0f);
  // 8) gated residual
  k_gated_res<<<ew_blocks(nQH), 256, 0, stream>>>(attno_f, q_f, rgate, x_f, x_bf, nQH);
  // 9) SwiGLU
  gemm(x_bf, gw_bf, gate_f, nullptr, nullptr,
       NQ_C, FFN_C, H_C, (long long)NQ_C * H_C, 0, (long long)NQ_C * FFN_C, 0, 1.0f);
  gemm(x_bf, uw_bf, up_f, nullptr, nullptr,
       NQ_C, FFN_C, H_C, (long long)NQ_C * H_C, 0, (long long)NQ_C * FFN_C, 0, 1.0f);
  k_swiglu<<<ew_blocks(nQF), 256, 0, stream>>>(gate_f, up_f, prod_bf, nQF);
  gemm(prod_bf, dw_bf, ffn_f, nullptr, nullptr,
       NQ_C, H_C, FFN_C, (long long)NQ_C * FFN_C, 0, (long long)NQ_C * H_C, 0, 1.0f);
  k_ffn_res<<<ew_blocks(nQH), 256, 0, stream>>>(ffn_f, x_f, frg, x2_f, nQH);
  // 10) RMSNorm
  {
    dim3 g(NQ_C, L_C);
    k_rmsnorm<<<g, 256, 0, stream>>>(x2_f, lnw, xn_bf);
  }
  // 11) frozen KV projections   [NQ,KV] per layer
  gemm(xn_bf, kproj_bf, kout_f, nullptr, nullptr,
       NQ_C, KV_C, H_C, (long long)NQ_C * H_C, (long long)KV_C * H_C,
       (long long)NQ_C * KV_C, 0, 1.0f);
  gemm(xn_bf, vproj_bf, vout_f, nullptr, nullptr,
       NQ_C, KV_C, H_C, (long long)NQ_C * H_C, (long long)KV_C * H_C,
       (long long)NQ_C * KV_C, 0, 1.0f);
  // 12) pack to [L,KVH,NQ,HD] x2
  k_pack_out<<<ew_blocks((size_t)L_C * KVH_C * NQ_C * HD_C), 256, 0, stream>>>(kout_f, vout_f, out);
}
